// FullAttentionBlock_62612033241793
// MI455X (gfx1250) — matrix operand order repaired and checked by reading, not compiled, not run on hardware
//
#include <hip/hip_runtime.h>
#include <hip/hip_bf16.h>
#include <math.h>

// ---------------------------------------------------------------------------
// MI455X (gfx1250) implementation: all GEMMs via v_wmma_f32_16x16x32_bf16
// (bf16 inputs, fp32 accumulate); attention V-tile staged with the Tensor
// Data Mover (TENSOR_LOAD_TO_LDS + s_wait_tensorcnt). ~130 GFLOP total,
// working set < 192MB L2. Workspace layout (needs >= 76 MB):
//   [ 0MB) qg   bf16 [4096,4096]  q_and_gate projection
//   [32MB) kraw bf16 [4096, 512]
//   [36MB) vraw bf16 [4096, 512]
//   [40MB) qb   bf16 [B,H,S,D]    normed+roped+scaled q
//   [56MB) kb   bf16 [B,HKV,S,D]  normed+roped k
//   [60MB) ao   bf16 [4096,2048]  gated attention output
// ---------------------------------------------------------------------------

typedef __bf16 bf16;
typedef __attribute__((ext_vector_type(16))) __bf16 v16bf;
typedef __attribute__((ext_vector_type(8)))  float  v8f;
typedef __attribute__((ext_vector_type(4)))  unsigned int u32x4;
typedef __attribute__((ext_vector_type(8)))  int i32x8;
typedef __attribute__((ext_vector_type(4)))  int i32x4;

#define HIDN 1024
#define NB   2
#define NS   2048
#define NH   8
#define NKV  2
#define HD   256
#define GSZ  (NH / NKV)
#define RDIM 64

// A-fragment k base for 16-bit 16x32 A matrix (wave32 layout per ISA 7.12.2)
__device__ __forceinline__ int a_kbase(int j, int hi) {
    return ((j & 3) * 2) + ((j >> 2) * 16) + hi * 8;
}

// ---------------------------------------------------------------------------
// TDM: issue a 2D bf16 tile load Global -> LDS via TENSOR_LOAD_TO_LDS.
// Descriptor bitfields per CDNA5 ISA 8.3/8.4 (D# group0 / group1).
// tile: tile_d1 rows x tile_d0 elems; global row stride = row_stride elems;
// LDS destination is contiguous (tile_d1 * tile_d0 * 2 bytes).
// ---------------------------------------------------------------------------
__device__ __forceinline__ void tdm_load_2d_bf16(
    unsigned int lds_off, const void* gptr,
    unsigned int tile_d0, unsigned int tile_d1,
    unsigned int tensor_d0, unsigned int tensor_d1,
    unsigned long long row_stride)
{
    unsigned long long ga = (unsigned long long)(uintptr_t)gptr;
    u32x4 g0;
    g0[0] = 1u;                                    // count=1 valid user D#
    g0[1] = lds_off;                               // lds_addr (bytes)
    g0[2] = (unsigned int)(ga & 0xffffffffu);      // global_addr[31:0]
    g0[3] = (unsigned int)((ga >> 32) & 0x01ffffffu) | (2u << 30);  // type=2
    i32x8 g1;
    g1[0] = (int)(1u << 16);                       // data_size=1 (2 bytes)
    g1[1] = (int)((tensor_d0 & 0xffffu) << 16);    // tensor_dim0[15:0]
    g1[2] = (int)(((tensor_d0 >> 16) & 0xffffu) | ((tensor_d1 & 0xffffu) << 16));
    g1[3] = (int)(((tensor_d1 >> 16) & 0xffffu) | ((tile_d0 & 0xffffu) << 16));
    g1[4] = (int)(tile_d1 & 0xffffu);              // tile_dim1 (tile_dim2=0)
    g1[5] = (int)(unsigned int)(row_stride & 0xffffffffull);   // dim0 stride
    g1[6] = (int)(unsigned int)((row_stride >> 32) & 0xffffull);
    g1[7] = 0;                                     // tensor_dim1_stride unused
    i32x4 gz = {0, 0, 0, 0};
#if __has_include(<hip/amd_detail/amd_gfx1250_TDM.h>)
    i32x8 gz8 = {0, 0, 0, 0, 0, 0, 0, 0};
    __builtin_amdgcn_tensor_load_to_lds(g0, g1, gz, gz, gz8, 0);   // 6-arg
#else
    __builtin_amdgcn_tensor_load_to_lds(g0, g1, gz, gz, 0);        // 5-arg
#endif
}

// ---------------------------------------------------------------------------
// Generic tiled GEMM: C[M,N] = A[M,K] @ W[N,K]^T
// Block tile 128x128, K step 32, 8 waves (256 thr), wave tile 32x64.
// A_BF16: A is bf16 (else fp32, converted). OUT_F32: write fp32 (else bf16).
// All dims assumed multiples of tile sizes (true for this problem).
// ---------------------------------------------------------------------------
template <bool A_BF16, bool OUT_F32>
__global__ __launch_bounds__(256) void gemm_bf16wmma(
    const void* __restrict__ Ap, const float* __restrict__ W,
    void* __restrict__ Outp, int M, int N, int K)
{
    __shared__ alignas(16) bf16 As[128 * 32];  // [m][k], ld 32
    __shared__ alignas(16) bf16 Bs[32 * 128];  // [k][n], ld 128

    const int tid  = threadIdx.x;
    const int lane = tid & 31;
    const int wave = tid >> 5;
    const int hi   = lane >> 4;
    const int l15  = lane & 15;

    const int m0 = blockIdx.y * 128;
    const int n0 = blockIdx.x * 128;
    const int wm = (wave & 3) * 32;   // wave M offset inside block tile
    const int wn = (wave >> 2) * 64;  // wave N offset inside block tile

    v8f acc[2][4];
#pragma unroll
    for (int mi = 0; mi < 2; ++mi)
#pragma unroll
        for (int ni = 0; ni < 4; ++ni)
#pragma unroll
            for (int i = 0; i < 8; ++i) acc[mi][ni][i] = 0.0f;

    for (int k0 = 0; k0 < K; k0 += 32) {
        __syncthreads();
        // ---- stage A tile (128x32) into LDS as bf16, [m][k] ----
#pragma unroll
        for (int it = 0; it < 4; ++it) {
            int s   = tid + 256 * it;      // 0..1023 slots of 4 elems
            int row = s >> 3;
            int c4  = (s & 7) * 4;
            if (A_BF16) {
                const bf16* A = (const bf16*)Ap;
#pragma unroll
                for (int q = 0; q < 4; ++q)
                    As[row * 32 + c4 + q] = A[(size_t)(m0 + row) * K + k0 + c4 + q];
            } else {
                const float* A = (const float*)Ap;
#pragma unroll
                for (int q = 0; q < 4; ++q)
                    As[row * 32 + c4 + q] = (bf16)A[(size_t)(m0 + row) * K + k0 + c4 + q];
            }
        }
        // ---- stage B tile: W[n][k] -> Bs[k][n] (transposed in LDS) ----
#pragma unroll
        for (int it = 0; it < 4; ++it) {
            int s   = tid + 256 * it;
            int nrw = s >> 3;
            int c4  = (s & 7) * 4;
#pragma unroll
            for (int q = 0; q < 4; ++q)
                Bs[(c4 + q) * 128 + nrw] = (bf16)W[(size_t)(n0 + nrw) * K + k0 + c4 + q];
        }
        // prefetch next K tile (global_prefetch_b8)
        if (k0 + 32 < K) {
            __builtin_prefetch(&W[(size_t)(n0 + (tid >> 1)) * K + k0 + 32], 0, 1);
        }
        __syncthreads();

        // ---- fragments + WMMA ----
        v16bf afrag[2];
#pragma unroll
        for (int mi = 0; mi < 2; ++mi) {
            int m = wm + 16 * mi + l15;
#pragma unroll
            for (int j = 0; j < 8; ++j) {
                int kk = a_kbase(j, hi);
                afrag[mi][2 * j]     = As[m * 32 + kk];
                afrag[mi][2 * j + 1] = As[m * 32 + kk + 1];
            }
        }
#pragma unroll
        for (int ni = 0; ni < 4; ++ni) {
            v16bf bfrag;
#pragma unroll
            for (int j = 0; j < 8; ++j) {   // lane = n column, packed k pair (A-frag k order)
                int kb = a_kbase(j, hi);
                bfrag[2 * j]     = Bs[kb * 128 + wn + 16 * ni + l15];
                bfrag[2 * j + 1] = Bs[(kb + 1) * 128 + wn + 16 * ni + l15];
            }
#pragma unroll
            for (int mi = 0; mi < 2; ++mi)
                acc[mi][ni] = __builtin_amdgcn_wmma_f32_16x16x32_bf16(
                    false, afrag[mi], false, bfrag, (short)0, acc[mi][ni], false, false);
        }
    }

    // ---- epilogue: C layout lane=N (l15), VGPR i -> M = i + 8*hi ----
#pragma unroll
    for (int mi = 0; mi < 2; ++mi)
#pragma unroll
        for (int ni = 0; ni < 4; ++ni)
#pragma unroll
            for (int i = 0; i < 8; ++i) {
                int gm  = m0 + wm + 16 * mi + i + 8 * hi;
                int gn  = n0 + wn + 16 * ni + l15;
                float v = acc[mi][ni][i];
                if (OUT_F32)
                    ((float*)Outp)[(size_t)gm * N + gn] = v;
                else
                    ((bf16*)Outp)[(size_t)gm * N + gn] = (bf16)v;
            }
}

// ---------------------------------------------------------------------------
// RMSNorm + partial RoPE for q and k. One wave per (token, head) row.
// Lane owns d = lane + 32e (e=0..7); RoPE pairs (d, d+32) are lane-local.
// Folds the 1/sqrt(D) score scale into q.
// ---------------------------------------------------------------------------
__global__ __launch_bounds__(256) void qk_prep(
    const bf16* __restrict__ qg, const bf16* __restrict__ kraw,
    const int* __restrict__ pos_ids,
    const float* __restrict__ qw, const float* __restrict__ kw,
    bf16* __restrict__ qb, bf16* __restrict__ kb)
{
    const int wave = threadIdx.x >> 5;
    const int lane = threadIdx.x & 31;
    const int r    = blockIdx.x * 8 + wave;      // (token, head) row id
    const int token = r / (NH + NKV);
    const int hh    = r % (NH + NKV);
    const int b = token / NS, s = token % NS;
    const bool isq = (hh < NH);
    const int head = isq ? hh : hh - NH;

    const bf16* src = isq ? (qg + (size_t)token * (2 * NH * HD) + head * HD)
                          : (kraw + (size_t)token * (NKV * HD) + head * HD);
    const float* w = isq ? qw : kw;

    float x[8];
    float ss = 0.0f;
#pragma unroll
    for (int e = 0; e < 8; ++e) {
        x[e] = (float)src[lane + 32 * e];
        ss += x[e] * x[e];
    }
#pragma unroll
    for (int off = 16; off > 0; off >>= 1) ss += __shfl_xor(ss, off, 32);
    const float inv = rsqrtf(ss * (1.0f / HD) + 1e-6f);

    float y[8];
#pragma unroll
    for (int e = 0; e < 8; ++e) y[e] = w[lane + 32 * e] * x[e] * inv;

    // RoPE on first RDIM=64 dims: pair (lane, lane+32) = (y[0], y[1])
    const float pos  = (float)pos_ids[s];
    const float invf = expf(-(2.0f * (float)lane / (float)RDIM) * logf(10000000.0f));
    const float ang  = pos * invf;
    const float c = cosf(ang), sn = sinf(ang);
    const float y0 = y[0] * c - y[1] * sn;
    const float y1 = y[1] * c + y[0] * sn;
    y[0] = y0; y[1] = y1;

    const float scale = isq ? 0.0625f : 1.0f;  // 1/sqrt(256) folded into q
    bf16* dst = isq ? (qb + ((size_t)(b * NH + head) * NS + s) * HD)
                    : (kb + ((size_t)(b * NKV + head) * NS + s) * HD);
#pragma unroll
    for (int e = 0; e < 8; ++e) dst[lane + 32 * e] = (bf16)(y[e] * scale);
}

// ---------------------------------------------------------------------------
// Flash-style GQA attention + SiLU(gate) epilogue.
// Block = 128 thr (4 waves); each wave owns 16 queries; keys in chunks of 32.
// V chunk DMA'd to LDS by the Tensor Data Mover (wave 0 issues, TENSORcnt);
// K chunk staged TRANSPOSED in LDS ([d][key]) manually (TDM can't transpose),
// overlapped with the TDM transfer. Online softmax with shfl width-16 row
// reductions (matching the C-fragment halves).
// ---------------------------------------------------------------------------
__global__ __launch_bounds__(128) void attn_kernel(
    const bf16* __restrict__ qb, const bf16* __restrict__ kb,
    const bf16* __restrict__ vraw, const bf16* __restrict__ qg,
    bf16* __restrict__ ao)
{
    __shared__ alignas(16) bf16 Kt[HD * 32];      // [d][key], ld 32   (16 KB)
    __shared__ alignas(16) bf16 Vt[32 * HD];      // [key][d], ld 256  (16 KB)
    __shared__ alignas(16) bf16 Pt[4 * 16 * 32];  // per-wave P tiles   (4 KB)

    const int tid  = threadIdx.x;
    const int lane = tid & 31;
    const int wave = tid >> 5;
    const int hi   = lane >> 4;
    const int l15  = lane & 15;

    const int qt  = blockIdx.x;            // query tile (64 q / block)
    const int bh  = blockIdx.y;            // b * NH + hq
    const int b   = bh / NH;
    const int hq  = bh % NH;
    const int hkv = hq / GSZ;
    const int q0  = qt * 64 + wave * 16;   // wave's first query row in S

    const bf16* qbase = qb + (size_t)(b * NH + hq) * NS * HD;
    const bf16* kbase = kb + (size_t)(b * NKV + hkv) * NS * HD;
    const bf16* vbase = vraw + (size_t)b * NS * (NKV * HD) + hkv * HD;

    const unsigned int vt_lds = (unsigned int)(uintptr_t)&Vt[0];

    // Q fragments for all 8 head-dim chunks (pre-scaled by 1/sqrt(D))
    v16bf qf[8];
#pragma unroll
    for (int kk = 0; kk < 8; ++kk)
#pragma unroll
        for (int j = 0; j < 8; ++j) {
            int kd = kk * 32 + a_kbase(j, hi);
            qf[kk][2 * j]     = qbase[(size_t)(q0 + l15) * HD + kd];
            qf[kk][2 * j + 1] = qbase[(size_t)(q0 + l15) * HD + kd + 1];
        }

    v8f O[16];
#pragma unroll
    for (int t = 0; t < 16; ++t)
#pragma unroll
        for (int i = 0; i < 8; ++i) O[t][i] = 0.0f;
    float rmax[8], rsum[8];
#pragma unroll
    for (int i = 0; i < 8; ++i) { rmax[i] = -1e30f; rsum[i] = 0.0f; }

    for (int key0 = 0; key0 < NS; key0 += 32) {
        __syncthreads();
        // wave 0: kick off TDM for V chunk (32 rows x 256 bf16, row stride 512)
        if (wave == 0) {
            tdm_load_2d_bf16(vt_lds, vbase + (size_t)key0 * (NKV * HD),
                             /*tile_d0=*/HD, /*tile_d1=*/32,
                             /*tensor_d0=*/NKV * HD, /*tensor_d1=*/NB * NS,
                             /*row_stride=*/NKV * HD);
        }
        // all waves: stage K chunk transposed ([d][key]) while TDM runs
        for (int idx = tid; idx < 32 * HD; idx += 128) {
            int key = idx >> 8;
            int d   = idx & 255;
            Kt[d * 32 + key] = kbase[(size_t)(key0 + key) * HD + d];
        }
        if (wave == 0) __builtin_amdgcn_s_wait_tensorcnt(0);
        __syncthreads();

        // scores 16x32 = Q @ K^T (8 k-chunks of 32 per 16-key tile)
        v8f sc[2];
#pragma unroll
        for (int c = 0; c < 2; ++c) {
#pragma unroll
            for (int i = 0; i < 8; ++i) sc[c][i] = 0.0f;
#pragma unroll
            for (int kk = 0; kk < 8; ++kk) {
                v16bf kf;
#pragma unroll
                for (int j = 0; j < 8; ++j) {   // lane = key column, packed d pair
                    int kb = a_kbase(j, hi);
                    kf[2 * j]     = Kt[(kk * 32 + kb) * 32 + c * 16 + l15];
                    kf[2 * j + 1] = Kt[(kk * 32 + kb + 1) * 32 + c * 16 + l15];
                }
                sc[c] = __builtin_amdgcn_wmma_f32_16x16x32_bf16(
                    false, qf[kk], false, kf, (short)0, sc[c], false, false);
            }
        }

        // online softmax: rows i+8*hi live across the 16 lanes of each half
        float f[8];
#pragma unroll
        for (int i = 0; i < 8; ++i) {
            float m = fmaxf(sc[0][i], sc[1][i]);
#pragma unroll
            for (int off = 8; off > 0; off >>= 1) m = fmaxf(m, __shfl_xor(m, off, 16));
            float nm = fmaxf(rmax[i], m);
            f[i]     = __expf(rmax[i] - nm);
            rmax[i]  = nm;
            float p0 = __expf(sc[0][i] - nm);
            float p1 = __expf(sc[1][i] - nm);
            sc[0][i] = p0; sc[1][i] = p1;
            float ps = p0 + p1;
#pragma unroll
            for (int off = 8; off > 0; off >>= 1) ps += __shfl_xor(ps, off, 16);
            rsum[i] = rsum[i] * f[i] + ps;
        }
#pragma unroll
        for (int t = 0; t < 16; ++t)
#pragma unroll
            for (int i = 0; i < 8; ++i) O[t][i] *= f[i];

        // route P through LDS: C-frag layout -> A-frag layout
#pragma unroll
        for (int c = 0; c < 2; ++c)
#pragma unroll
            for (int i = 0; i < 8; ++i)
                Pt[wave * 512 + (i + 8 * hi) * 32 + c * 16 + l15] = (bf16)sc[c][i];
        __syncthreads();

        v16bf pf;
#pragma unroll
        for (int j = 0; j < 8; ++j) {
            int kd = a_kbase(j, hi);
            pf[2 * j]     = Pt[wave * 512 + l15 * 32 + kd];
            pf[2 * j + 1] = Pt[wave * 512 + l15 * 32 + kd + 1];
        }
        // O += P @ V (16 n-tiles cover D=256)
#pragma unroll
        for (int t = 0; t < 16; ++t) {
            v16bf vf;
#pragma unroll
            for (int j = 0; j < 8; ++j) {  // lane = d column, packed key pair
                int kb = a_kbase(j, hi);
                vf[2 * j]     = Vt[kb * HD + t * 16 + l15];
                vf[2 * j + 1] = Vt[(kb + 1) * HD + t * 16 + l15];
            }
            O[t] = __builtin_amdgcn_wmma_f32_16x16x32_bf16(
                false, pf, false, vf, (short)0, O[t], false, false);
        }
    }

    // epilogue: /rowsum, SiLU(gate), write bf16 [token][hq*HD + d]
    float rinv[8];
#pragma unroll
    for (int i = 0; i < 8; ++i) rinv[i] = 1.0f / rsum[i];

#pragma unroll
    for (int t = 0; t < 16; ++t)
#pragma unroll
        for (int i = 0; i < 8; ++i) {
            int m     = i + 8 * hi;
            int token = b * NS + q0 + m;
            int d     = t * 16 + l15;
            float g    = (float)qg[(size_t)token * (2 * NH * HD) + NH * HD + hq * HD + d];
            float silu = g / (1.0f + __expf(-g));
            float o    = O[t][i] * rinv[i];
            ao[(size_t)token * (NH * HD) + hq * HD + d] = (bf16)(o * silu);
        }
}

// ---------------------------------------------------------------------------
extern "C" void kernel_launch(void* const* d_in, const int* in_sizes, int n_in,
                              void* d_out, int out_size, void* d_ws, size_t ws_size,
                              hipStream_t stream)
{
    const float* hidden = (const float*)d_in[0];
    const int*   pos    = (const int*)d_in[1];
    const float* Wq     = (const float*)d_in[2];
    const float* Wk     = (const float*)d_in[3];
    const float* Wv     = (const float*)d_in[4];
    const float* Wo     = (const float*)d_in[5];
    const float* qw     = (const float*)d_in[6];
    const float* kw     = (const float*)d_in[7];

    char* ws  = (char*)d_ws;
    bf16* qg   = (bf16*)(ws);
    bf16* kraw = (bf16*)(ws + (size_t)32 * 1024 * 1024);
    bf16* vraw = (bf16*)(ws + (size_t)36 * 1024 * 1024);
    bf16* qb   = (bf16*)(ws + (size_t)40 * 1024 * 1024);
    bf16* kb   = (bf16*)(ws + (size_t)56 * 1024 * 1024);
    bf16* ao   = (bf16*)(ws + (size_t)60 * 1024 * 1024);

    const int M = NB * NS;  // 4096 tokens

    dim3 blk(256);
    // q_and_gate / k / v projections (fp32 in, bf16 out)
    gemm_bf16wmma<false, false><<<dim3((2 * NH * HD) / 128, M / 128), blk, 0, stream>>>(
        hidden, Wq, qg, M, 2 * NH * HD, HIDN);
    gemm_bf16wmma<false, false><<<dim3((NKV * HD) / 128, M / 128), blk, 0, stream>>>(
        hidden, Wk, kraw, M, NKV * HD, HIDN);
    gemm_bf16wmma<false, false><<<dim3((NKV * HD) / 128, M / 128), blk, 0, stream>>>(
        hidden, Wv, vraw, M, NKV * HD, HIDN);

    // RMSNorm + RoPE (+ fold score scale into q)
    qk_prep<<<dim3(M * (NH + NKV) / 8), blk, 0, stream>>>(qg, kraw, pos, qw, kw, qb, kb);

    // attention + SiLU gating (TDM-staged V)
    attn_kernel<<<dim3(NS / 64, NB * NH), dim3(128), 0, stream>>>(qb, kb, vraw, qg, ao);

    // output projection (bf16 in, fp32 out)
    gemm_bf16wmma<true, true><<<dim3(HIDN / 128, M / 128), blk, 0, stream>>>(
        ao, Wo, d_out, M, HIDN, NH * HD);
}
